// CausalSelfAttentionEx_62740882260809
// MI455X (gfx1250) — matrix operand order repaired and checked by reading, not compiled, not run on hardware
//
#include <hip/hip_runtime.h>

typedef __attribute__((ext_vector_type(16))) __bf16 v16bf;
typedef __attribute__((ext_vector_type(8)))  float  v8f;

// ---------------------------------------------------------------------------
// Fragment loaders for V_WMMA_F32_16X16X32_BF16 (wave32).
//
// A-matrix 16x32 (MxK), contiguous-K order used by this code (B uses the same):
//   lane L: row = L&15, half = L>>4
//   elems e -> K = half*16 + e  (16 contiguous K values)
__device__ __forceinline__ v16bf load_a_frag(const __bf16* __restrict__ A,
                                             int lda, int m0, int kb, int lane) {
  const int hf = lane >> 4, rw = lane & 15;
  const __bf16* p = A + (size_t)(m0 + rw) * lda + kb + hf * 16;
  v16bf f;
#pragma unroll
  for (int j = 0; j < 16; ++j) f[j] = p[j];
  return f;
}

// B-matrix 32x16 (KxN), same contiguous-K order as A:
//   lane L: col = L&15, elems e: K = (L>>4)*16 + e  (16 contiguous K values)
// Source operand must be laid out [N][K] (n-th row contiguous in k).
__device__ __forceinline__ v16bf load_b_frag(const __bf16* __restrict__ W,
                                             int ldw, int n0, int kb, int lane) {
  const int hf = lane >> 4, cl = lane & 15;
  const __bf16* p = W + (size_t)(n0 + cl) * ldw + kb + hf * 16;
  v16bf f;
#pragma unroll
  for (int j = 0; j < 16; ++j) f[j] = p[j];
  return f;
}

// ---------------------------------------------------------------------------
__global__ void cvt_f32_bf16(const float* __restrict__ s, __bf16* __restrict__ d, int n) {
  int i = blockIdx.x * blockDim.x + threadIdx.x;
  if (i < n) d[i] = (__bf16)s[i];
}

// ---------------------------------------------------------------------------
// out[m,n] = sum_k A[m,k] * W[n,k] + bias[n]
// One wave computes a 32x64 tile: 8 wmma per K-step off 6 b128-run loads.
// All fragments for a K-step are loaded before the WMMAs so load clauses
// overlap WMMA execution instead of draining loadcnt to 0 each time.
// Columns >= vcol are stored transposed into VT (feature-major) instead of Out.
template <bool F32OUT>
__global__ __launch_bounds__(256) void gemm_bias_wmma(
    const __bf16* __restrict__ A, const __bf16* __restrict__ W,
    const float* __restrict__ bias, void* __restrict__ Out,
    __bf16* __restrict__ VT, int vcol,
    int M, int N, int K) {
  const int wv = threadIdx.x >> 5, lane = threadIdx.x & 31;
  const int gw = blockIdx.x * (blockDim.x >> 5) + wv;
  const int n64 = N >> 6;
  const int mt = gw / n64, nt = gw - mt * n64;
  if (mt >= (M >> 5)) return;                 // wave-uniform: EXEC stays all-1s
  const int m0 = mt << 5, n0 = nt << 6;

  v8f acc[2][4] = {{{}, {}, {}, {}}, {{}, {}, {}, {}}};
  for (int kb = 0; kb < K; kb += 32) {
    v16bf a0 = load_a_frag(A, K, m0,      kb, lane);
    v16bf a1 = load_a_frag(A, K, m0 + 16, kb, lane);
    v16bf b[4];
#pragma unroll
    for (int t = 0; t < 4; ++t) b[t] = load_b_frag(W, K, n0 + t * 16, kb, lane);
#pragma unroll
    for (int t = 0; t < 4; ++t) {
      acc[0][t] = __builtin_amdgcn_wmma_f32_16x16x32_bf16(
          false, a0, false, b[t], (short)0, acc[0][t], false, false);
      acc[1][t] = __builtin_amdgcn_wmma_f32_16x16x32_bf16(
          false, a1, false, b[t], (short)0, acc[1][t], false, false);
    }
  }
  // C/D striping: VGPR g -> row = half*8 + g, col = lane&15
  const int hf = lane >> 4, cl = lane & 15;
  const bool toVT = (VT != nullptr) && (n0 >= vcol);   // wave-uniform (vcol%64==0)
#pragma unroll
  for (int mi = 0; mi < 2; ++mi) {
#pragma unroll
    for (int t = 0; t < 4; ++t) {
#pragma unroll
      for (int g = 0; g < 8; ++g) {
        const int r = m0 + mi * 16 + hf * 8 + g;
        const int c = n0 + t * 16 + cl;
        const float v = acc[mi][t][g] + bias[c];
        if (toVT)        VT[(size_t)(c - vcol) * M + r] = (__bf16)v;  // [feat][tok]
        else if (F32OUT) ((float*)Out)[(size_t)r * N + c] = v;
        else             ((__bf16*)Out)[(size_t)r * N + c] = (__bf16)v;
      }
    }
  }
}

// ---------------------------------------------------------------------------
// Flash attention: one wave handles one (b, h, 16-query-row tile).
// qkv: [B*T, 3C] bf16 (only q|k slices read);  vT: [C][B*T] bf16 (feature-major).
__global__ __launch_bounds__(256) void attn_wmma(
    const __bf16* __restrict__ qkv, const __bf16* __restrict__ vT,
    __bf16* __restrict__ y1) {
  __shared__ __bf16 pbuf[8][16 * 32];         // wave-private P staging tile
  const int T = 2048, C = 1024, C3 = 3072, BT = 4096;
  const int wv = threadIdx.x >> 5, lane = threadIdx.x & 31;
  const int gw = blockIdx.x * 8 + wv;
  if (gw >= 32 * 128) return;
  const int bh = gw >> 7, qt = gw & 127;
  const int b = bh >> 4, h = bh & 15;
  const int brow = b * T, q0 = qt * 16;
  const int hf = lane >> 4, cl = lane & 15;
  const float scale = 0.03125f;               // 1/sqrt(C), matches reference

  const __bf16* qp = qkv + (size_t)(h * 64);
  const __bf16* kp = qkv + (size_t)(C + h * 64);
  const __bf16* vp = vT + (size_t)(h * 64) * BT;   // head slice, [d][token]

  v16bf qf[2];
#pragma unroll
  for (int c = 0; c < 2; ++c) qf[c] = load_a_frag(qp, C3, brow + q0, c * 32, lane);

  // Ones-column B fragment: folds the softmax row-sum into the PV WMMA.
  v16bf onef;
#pragma unroll
  for (int e = 0; e < 16; ++e) onef[e] = (cl == 0) ? (__bf16)1.0f : (__bf16)0.0f;

  v8f o[4] = {{}, {}, {}, {}};
  v8f o4 = {};                                // col0 accumulates l = sum exp
  float mrow[8];
#pragma unroll
  for (int g = 0; g < 8; ++g) mrow[g] = -1e30f;

  const int nkb = (q0 + 16 + 31) >> 5;        // causal: only keys <= q0+15
  for (int ib = 0; ib < nkb; ++ib) {
    const int kb = ib * 32;
    // Unconditional prefetch of next block (stays inside workspace even past
    // the causal frontier; avoids a divergent EXEC-masked region per iter).
    __builtin_prefetch(kp + (size_t)(brow + kb + 32 + cl) * C3, 0, 1);
    __builtin_prefetch(vp + (size_t)(cl)*BT + brow + kb + 32, 0, 1);

    // ---- load ALL K fragments first, then issue the 4 score WMMAs
    v16bf kf[2][2];
#pragma unroll
    for (int st = 0; st < 2; ++st)
#pragma unroll
      for (int c = 0; c < 2; ++c)
        kf[st][c] = load_b_frag(kp, C3, brow + kb + st * 16, c * 32, lane);

    v8f s[2];
#pragma unroll
    for (int st = 0; st < 2; ++st) {
      v8f sa = {};
      sa = __builtin_amdgcn_wmma_f32_16x16x32_bf16(
          false, qf[0], false, kf[st][0], (short)0, sa, false, false);
      sa = __builtin_amdgcn_wmma_f32_16x16x32_bf16(
          false, qf[1], false, kf[st][1], (short)0, sa, false, false);
      s[st] = sa;
    }

    // ---- prefetch V fragments for this block (independent of softmax chain)
    v16bf vf[4];
#pragma unroll
    for (int t = 0; t < 4; ++t)
      vf[t] = load_b_frag(vp, BT, t * 16, brow + kb, lane);

    // ---- online softmax over the 32-key block (max via shuffles only;
    //      the row-sum rides along in the o4 WMMA accumulator)
    float p0[8], p1[8];
#pragma unroll
    for (int g = 0; g < 8; ++g) {
      const int qr = q0 + hf * 8 + g;
      float a0 = s[0][g] * scale;
      float a1 = s[1][g] * scale;
      if (kb + cl > qr)      a0 = -1e30f;     // causal mask
      if (kb + 16 + cl > qr) a1 = -1e30f;
      float bm = fmaxf(a0, a1);
#pragma unroll
      for (int off = 8; off >= 1; off >>= 1) bm = fmaxf(bm, __shfl_xor(bm, off));
      const float mn  = fmaxf(mrow[g], bm);
      const float fac = __expf(mrow[g] - mn);
      mrow[g] = mn;
      p0[g] = __expf(a0 - mn);
      p1[g] = __expf(a1 - mn);
#pragma unroll
      for (int t = 0; t < 4; ++t) o[t][g] *= fac;
      o4[g] *= fac;
    }

    // ---- re-stripe P from C/D layout to A layout through LDS
#pragma unroll
    for (int g = 0; g < 8; ++g) {
      pbuf[wv][(hf * 8 + g) * 32 + cl]      = (__bf16)p0[g];
      pbuf[wv][(hf * 8 + g) * 32 + 16 + cl] = (__bf16)p1[g];
    }
    asm volatile("s_wait_dscnt 0" ::: "memory");    // cross-lane LDS hazard
    v16bf pf;
#pragma unroll
    for (int e = 0; e < 16; ++e)
      pf[e] = pbuf[wv][cl * 32 + hf * 16 + e];

    // ---- O += P (16x32) x V (32x64), plus ones-column for l
#pragma unroll
    for (int t = 0; t < 4; ++t)
      o[t] = __builtin_amdgcn_wmma_f32_16x16x32_bf16(
          false, pf, false, vf[t], (short)0, o[t], false, false);
    o4 = __builtin_amdgcn_wmma_f32_16x16x32_bf16(
        false, pf, false, onef, (short)0, o4, false, false);
  }

  // ---- normalize (l lives in col0 of o4 -> broadcast) and store heads
#pragma unroll
  for (int g = 0; g < 8; ++g) {
    const float l   = __shfl(o4[g], hf * 16);   // col 0 of this half
    const float inv = 1.0f / l;
    const int r = brow + q0 + hf * 8 + g;
#pragma unroll
    for (int t = 0; t < 4; ++t)
      y1[(size_t)r * C + h * 64 + t * 16 + cl] = (__bf16)(o[t][g] * inv);
  }
}

// ---------------------------------------------------------------------------
extern "C" void kernel_launch(void* const* d_in, const int* in_sizes, int n_in,
                              void* d_out, int out_size, void* d_ws, size_t ws_size,
                              hipStream_t stream) {
  const float* x      = (const float*)d_in[0];
  // d_in[1] = mask (unused by reference)
  const float* W_attn = (const float*)d_in[2];
  const float* b_attn = (const float*)d_in[3];
  const float* W_proj = (const float*)d_in[4];
  const float* b_proj = (const float*)d_in[5];

  const int BT = 4096, C = 1024, C3 = 3072;

  char* ws = (char*)d_ws;
  __bf16* xb   = (__bf16*)(ws);                       //  8 MB: x bf16
  __bf16* Wab  = (__bf16*)(ws + 8388608);             //  6 MB: W_attn bf16
  __bf16* Wpb  = (__bf16*)(ws + 14680064);            //  2 MB: W_proj bf16
  __bf16* qkvb = (__bf16*)(ws + 16777216);            // 24 MB: q|k rows (v unused)
  __bf16* vTb  = (__bf16*)(ws + 41943040);            //  8 MB: v transposed [C][BT]
  __bf16* y1b  = (__bf16*)(ws + 50331648);            //  8 MB: attn out bf16

  cvt_f32_bf16<<<(BT * C + 255) / 256, 256, 0, stream>>>(x, xb, BT * C);
  cvt_f32_bf16<<<(C3 * C + 255) / 256, 256, 0, stream>>>(W_attn, Wab, C3 * C);
  cvt_f32_bf16<<<(C * C + 255) / 256, 256, 0, stream>>>(W_proj, Wpb, C * C);

  // qkv = x @ W_attn^T + b_attn (M=4096,N=3072,K=1024); V slice lands in vTb
  gemm_bias_wmma<false><<<(BT / 32) * (C3 / 64) / 8, 256, 0, stream>>>(
      xb, Wab, b_attn, qkvb, vTb, 2048, BT, C3, C);

  // flash attention over 32 (b,h) x 128 query tiles
  attn_wmma<<<512, 256, 0, stream>>>(qkvb, vTb, y1b);

  // out = y1 @ W_proj^T + b_proj (M=4096,N=1024,K=1024) -> fp32
  gemm_bias_wmma<true><<<(BT / 32) * (C / 64) / 8, 256, 0, stream>>>(
      y1b, Wpb, b_proj, d_out, nullptr, 1 << 30, BT, C, C);
}
